// Seq2SeqLSTMWithAttention_11115375362492
// MI455X (gfx1250) — matrix operand order repaired and checked by reading, compiled, not run on hardware
//
#include <hip/hip_runtime.h>
#include <hip/hip_bf16.h>
#include <math.h>

typedef __attribute__((ext_vector_type(16))) _Float16 v16h;
typedef __attribute__((ext_vector_type(8)))  _Float16 v8h;
typedef __attribute__((ext_vector_type(8)))  float    v8f;

// ---------------------------------------------------------------------------
// WMMA GEMM, all-f16 operands, f32 accumulate.
//   A  : f16 [M][lda]  row-major activations
//   BT : f16 [N][ldb]  TRANSPOSED weights (row n = column n of B), ldb = K
//   C  : f32 [M][ldc] = (accum? C:0) + A@B + bias
// One wave computes a 32x16 macro-tile (two 16x16 WMMA tiles sharing one B
// fragment -> halves B traffic). All fragment loads are 16B/32B vector loads:
//   A frag: two v8h (contiguous K runs {ka..ka+7},{ka+8..ka+15})
//   B frag: one v16h (contiguous K run kb..kb+15 of B^T row)
// Fragment k order (wave32): contiguous k = 16*half + s, same for A and B.
// ---------------------------------------------------------------------------
__global__ void wmma_gemm16(const _Float16* __restrict__ A, int lda,
                            const _Float16* __restrict__ BT, int ldb,
                            const float* __restrict__ bias,
                            float* __restrict__ C, int ldc,
                            int Mtiles32, int Ntiles, int K, int accumulate)
{
    int wave = (int)((blockIdx.x * blockDim.x + threadIdx.x) >> 5);
    if (wave >= Mtiles32 * Ntiles) return;          // wave-uniform
    int tm = wave / Ntiles;
    int tn = wave % Ntiles;

    int lane = (int)(threadIdx.x & 31);
    int half = lane >> 4;
    int l16  = lane & 15;

    const _Float16* __restrict__ Arow0 = A + (size_t)(tm * 32 + l16) * (size_t)lda;
    const _Float16* __restrict__ Arow1 = Arow0 + (size_t)16 * (size_t)lda;
    const _Float16* __restrict__ Brow  = BT + (size_t)(tn * 16 + l16) * (size_t)ldb;

    v8f acc0 = {}, acc1 = {};
    for (int k0 = 0; k0 < K; k0 += 32) {
        int ka = k0 + half * 16;
        int kb = k0 + half * 16;
        v16h b  = *(const v16h*)(Brow + kb);        // 32B contiguous
        v8h a0l = *(const v8h*)(Arow0 + ka);
        v8h a0h = *(const v8h*)(Arow0 + ka + 8);
        v8h a1l = *(const v8h*)(Arow1 + ka);
        v8h a1h = *(const v8h*)(Arow1 + ka + 8);
        if (k0 + 32 < K) __builtin_prefetch(Brow + kb + 32);
        v16h a0 = __builtin_shufflevector(a0l, a0h, 0,1,2,3,4,5,6,7,8,9,10,11,12,13,14,15);
        v16h a1 = __builtin_shufflevector(a1l, a1h, 0,1,2,3,4,5,6,7,8,9,10,11,12,13,14,15);
        acc0 = __builtin_amdgcn_wmma_f32_16x16x32_f16(false, a0, false, b, (short)0, acc0, false, false);
        acc1 = __builtin_amdgcn_wmma_f32_16x16x32_f16(false, a1, false, b, (short)0, acc1, false, false);
    }

    int bcol = tn * 16 + l16;
    float bv = bias ? bias[bcol] : 0.0f;
#pragma unroll
    for (int r = 0; r < 8; ++r) {
        int m0 = tm * 32 + r + half * 8;
        int m1 = m0 + 16;
        size_t i0 = (size_t)m0 * (size_t)ldc + bcol;
        size_t i1 = (size_t)m1 * (size_t)ldc + bcol;
        float v0 = acc0[r] + bv;
        float v1 = acc1[r] + bv;
        if (accumulate) { v0 += C[i0]; v1 += C[i1]; }
        C[i0] = v0;
        C[i1] = v1;
    }
}

// ---------------------------------------------------------------------------
// One-time weight prep: f32 [K][N] -> f16 transposed [N][K], LDS-tiled.
// ---------------------------------------------------------------------------
__global__ void transpose_to_f16(const float* __restrict__ in,
                                 _Float16* __restrict__ out, int K, int N)
{
    __shared__ _Float16 tile[32][33];
    int n0 = (int)blockIdx.x * 32, k0 = (int)blockIdx.y * 32;
    int tx = (int)threadIdx.x, ty = (int)threadIdx.y;      // (32, 8)
#pragma unroll
    for (int j = 0; j < 4; ++j)
        tile[ty + j * 8][tx] = (_Float16)in[(size_t)(k0 + ty + j * 8) * N + n0 + tx];
    __syncthreads();
#pragma unroll
    for (int j = 0; j < 4; ++j)
        out[(size_t)(n0 + ty + j * 8) * K + k0 + tx] = tile[tx][ty + j * 8];
}

// Gather embedding rows + convert to f16: out[r][0..E) = (f16)table[idx[r]][...]
__global__ void gather_rows_f16(const int* __restrict__ idx,
                                const float* __restrict__ table,
                                _Float16* __restrict__ out, int E, int total)
{
    int i = (int)(blockIdx.x * blockDim.x + threadIdx.x);
    if (i >= total) return;
    int r = i / E, e = i - r * E;
    out[i] = (_Float16)table[(size_t)idx[r] * E + e];
}

// ---------------------------------------------------------------------------
// Persistent encoder scan: one workgroup (32 waves), h f16 double-buffered in
// LDS, c f32 in LDS, 256 steps, one barrier per step. B operand comes from
// pre-transposed f16 enc_Wh^T (contiguous v16h loads); A from LDS via 16B
// ds_load_b128. Writes enc_hs in both f32 (for attention ctx) and f16 (GEMM A).
// ---------------------------------------------------------------------------
#define HPAD 520   // row stride (f16 / f32), multiple of 8 keeps 16B alignment

__global__ void encoder_scan(const float* __restrict__ zx,
                             const _Float16* __restrict__ WhT,  // [2048][512]
                             const float* __restrict__ bh,
                             float* __restrict__ enc_hs,
                             _Float16* __restrict__ enc_hs16,
                             _Float16* __restrict__ hstate16,
                             float* __restrict__ cstate,
                             int Ts)
{
    extern __shared__ char smem[];
    _Float16* hbuf0 = (_Float16*)smem;                  // 32*HPAD f16
    _Float16* hbuf1 = hbuf0 + 32 * HPAD;                // 32*HPAD f16
    float*    cbuf  = (float*)(hbuf1 + 32 * HPAD);      // 32*HPAD f32

    int tid = (int)threadIdx.x;
    for (int i = tid; i < 32 * HPAD; i += (int)blockDim.x) {
        hbuf0[i] = (_Float16)0.0f;
        hbuf1[i] = (_Float16)0.0f;
        cbuf[i]  = 0.0f;
    }
    __syncthreads();

    int w    = tid >> 5;
    int lane = tid & 31;
    int half = lane >> 4;
    int l16  = lane & 15;

    for (int t = 0; t < Ts; ++t) {
        _Float16* hr = (t & 1) ? hbuf1 : hbuf0;
        _Float16* hw = (t & 1) ? hbuf0 : hbuf1;

        for (int g2 = 0; g2 < 2; ++g2) {
            int G  = w * 2 + g2;                  // 64 groups: 2 M-tiles x 32 col-tiles
            int tm = G >> 5;
            int jt = G & 31;
            const _Float16* arow = hr + (tm * 16 + l16) * HPAD;

            v8f acc[4] = {};
            for (int k0 = 0; k0 < 512; k0 += 32) {
                int ka = k0 + half * 16;
                int kb = k0 + half * 16;
                v8h al = *(const v8h*)(arow + ka);
                v8h ah = *(const v8h*)(arow + ka + 8);
                v16h a = __builtin_shufflevector(al, ah, 0,1,2,3,4,5,6,7,8,9,10,11,12,13,14,15);
#pragma unroll
                for (int gate = 0; gate < 4; ++gate) {
                    int col = gate * 512 + jt * 16 + l16;
                    v16h b = *(const v16h*)(WhT + (size_t)col * 512 + kb);
                    acc[gate] = __builtin_amdgcn_wmma_f32_16x16x32_f16(
                                    false, a, false, b, (short)0, acc[gate],
                                    false, false);
                }
            }

            int jcol = jt * 16 + l16;
#pragma unroll
            for (int r = 0; r < 8; ++r) {
                int b = tm * 16 + r + half * 8;
                size_t zrow = ((size_t)b * (size_t)Ts + (size_t)t) * 2048;
                float zi = acc[0][r] + zx[zrow +    0 + jcol] + bh[   0 + jcol];
                float zf = acc[1][r] + zx[zrow +  512 + jcol] + bh[ 512 + jcol];
                float zg = acc[2][r] + zx[zrow + 1024 + jcol] + bh[1024 + jcol];
                float zo = acc[3][r] + zx[zrow + 1536 + jcol] + bh[1536 + jcol];
                float ii = 1.0f / (1.0f + __expf(-zi));
                float ff = 1.0f / (1.0f + __expf(-zf));
                float gg = tanhf(zg);
                float oo = 1.0f / (1.0f + __expf(-zo));
                float c  = ff * cbuf[b * HPAD + jcol] + ii * gg;
                cbuf[b * HPAD + jcol] = c;
                float h  = oo * tanhf(c);
                hw[b * HPAD + jcol] = (_Float16)h;
                size_t hsrow = ((size_t)b * (size_t)Ts + (size_t)t) * 512 + jcol;
                enc_hs[hsrow]   = h;
                enc_hs16[hsrow] = (_Float16)h;
                if (t == Ts - 1) {
                    hstate16[b * 512 + jcol] = (_Float16)h;
                    cstate[b * 512 + jcol]   = c;
                }
            }
        }
        __syncthreads();
    }
}

// ---------------------------------------------------------------------------
// Bahdanau energy: one wave per (b,s); wave32 shuffle reduction.
// ---------------------------------------------------------------------------
__global__ void attn_energy(const float* __restrict__ Wh_enc,
                            const float* __restrict__ hWs,
                            const float* __restrict__ v,
                            float* __restrict__ energy)
{
    int wave = (int)((blockIdx.x * blockDim.x + threadIdx.x) >> 5);
    int lane = (int)(threadIdx.x & 31);
    int b = wave >> 8;        // Ts = 256
    int s = wave & 255;
    if (b >= 32) return;
    const float* we = Wh_enc + ((size_t)b * 256 + s) * 512;
    const float* hw = hWs + (size_t)b * 512;
    float sum = 0.0f;
    for (int k = lane; k < 512; k += 32)
        sum += tanhf(we[k] + hw[k]) * v[k];
    for (int off = 16; off; off >>= 1)
        sum += __shfl_xor(sum, off, 32);
    if (lane == 0) energy[b * 256 + s] = sum;
}

// ---------------------------------------------------------------------------
// Softmax over Ts + ctx accumulation; ctx emitted in f16 straight into the
// GEMM A buffers: hc16[:,512:1024] and xbuf16[:,256:768].
// ---------------------------------------------------------------------------
__global__ void softmax_ctx(const float* __restrict__ energy,
                            const float* __restrict__ enc_hs,
                            _Float16* __restrict__ hc16,
                            _Float16* __restrict__ xbuf16)
{
    __shared__ float p[256];
    __shared__ float red[256];
    int b = (int)blockIdx.x;
    int s = (int)threadIdx.x;

    float e = energy[b * 256 + s];
    red[s] = e;
    __syncthreads();
    for (int off = 128; off; off >>= 1) {
        if (s < off) red[s] = fmaxf(red[s], red[s + off]);
        __syncthreads();
    }
    float mx = red[0];
    __syncthreads();

    float pe = __expf(e - mx);
    p[s] = pe;
    red[s] = pe;
    __syncthreads();
    for (int off = 128; off; off >>= 1) {
        if (s < off) red[s] += red[s + off];
        __syncthreads();
    }
    float inv = 1.0f / red[0];
    __syncthreads();

    for (int j = s; j < 512; j += 256) {
        float acc = 0.0f;
        for (int t2 = 0; t2 < 256; ++t2)
            acc += p[t2] * enc_hs[((size_t)b * 256 + t2) * 512 + j];
        acc *= inv;
        hc16[b * 1024 + 512 + j]  = (_Float16)acc;
        xbuf16[b * 768 + 256 + j] = (_Float16)acc;
    }
}

// x16[:, 0:256] = (f16) dec_embed[tgt[:, t]]
__global__ void build_x(const int* __restrict__ tgt, int t,
                        const float* __restrict__ dec_embed,
                        _Float16* __restrict__ xbuf16)
{
    int i = (int)(blockIdx.x * blockDim.x + threadIdx.x);
    if (i >= 32 * 256) return;
    int b = i >> 8, e = i & 255;
    int tok = tgt[b * 64 + t];
    xbuf16[b * 768 + e] = (_Float16)dec_embed[(size_t)tok * 256 + e];
}

// Decoder LSTM nonlinearity; h emitted in f16 into GEMM A buffers.
__global__ void dec_gates(const float* __restrict__ z,
                          _Float16* __restrict__ hstate16,
                          float* __restrict__ cstate,
                          _Float16* __restrict__ hc16)
{
    int i = (int)(blockIdx.x * blockDim.x + threadIdx.x);
    if (i >= 32 * 512) return;
    int b = i >> 9, j = i & 511;
    const float* zr = z + (size_t)b * 2048;
    float ii = 1.0f / (1.0f + __expf(-zr[j]));
    float ff = 1.0f / (1.0f + __expf(-zr[512 + j]));
    float gg = tanhf(zr[1024 + j]);
    float oo = 1.0f / (1.0f + __expf(-zr[1536 + j]));
    float c = ff * cstate[i] + ii * gg;
    cstate[i] = c;
    float h = oo * tanhf(c);
    hstate16[i] = (_Float16)h;
    hc16[b * 1024 + j] = (_Float16)h;
}

// Row-wise log-softmax over V=32000; writes d_out[:, t, :].
__global__ void log_softmax_out(const float* __restrict__ logits,
                                float* __restrict__ out, int t)
{
    __shared__ float red[256];
    int b = (int)blockIdx.x;
    int tid = (int)threadIdx.x;
    const float* lr = logits + (size_t)b * 32000;

    float mx = -1e30f;
    for (int v = tid; v < 32000; v += 256) mx = fmaxf(mx, lr[v]);
    red[tid] = mx;
    __syncthreads();
    for (int off = 128; off; off >>= 1) {
        if (tid < off) red[tid] = fmaxf(red[tid], red[tid + off]);
        __syncthreads();
    }
    mx = red[0];
    __syncthreads();

    float sum = 0.0f;
    for (int v = tid; v < 32000; v += 256) sum += __expf(lr[v] - mx);
    red[tid] = sum;
    __syncthreads();
    for (int off = 128; off; off >>= 1) {
        if (tid < off) red[tid] += red[tid + off];
        __syncthreads();
    }
    float lse = logf(red[0]) + mx;

    float* orow = out + ((size_t)b * 64 + t) * 32000;
    for (int v = tid; v < 32000; v += 256) orow[v] = lr[v] - lse;
}

// ---------------------------------------------------------------------------
extern "C" void kernel_launch(void* const* d_in, const int* in_sizes, int n_in,
                              void* d_out, int out_size, void* d_ws, size_t ws_size,
                              hipStream_t stream)
{
    (void)in_sizes; (void)n_in; (void)out_size; (void)ws_size;

    const int*   src       = (const int*)  d_in[0];   // [32,256]
    const int*   tgt       = (const int*)  d_in[1];   // [32,64]
    const float* enc_embed = (const float*)d_in[2];   // [32000,256]
    const float* enc_Wx    = (const float*)d_in[3];   // [256,2048]
    const float* enc_bx    = (const float*)d_in[4];
    const float* enc_Wh    = (const float*)d_in[5];   // [512,2048]
    const float* enc_bh    = (const float*)d_in[6];
    const float* dec_embed = (const float*)d_in[7];   // [32000,256]
    const float* attn_Wh   = (const float*)d_in[8];   // [512,512]
    const float* attn_Ws   = (const float*)d_in[9];   // [512,512]
    const float* attn_v    = (const float*)d_in[10];  // [512]
    const float* dec_Wx    = (const float*)d_in[11];  // [768,2048]
    const float* dec_bx    = (const float*)d_in[12];
    const float* dec_Wh    = (const float*)d_in[13];  // [512,2048]
    const float* dec_bh    = (const float*)d_in[14];
    const float* out_W     = (const float*)d_in[15];  // [1024,32000]
    const float* out_b     = (const float*)d_in[16];
    float* out = (float*)d_out;                       // [32,64,32000]

    // ---- workspace carve-up: f32 region then f16 region (~184 MB) ----
    float* zx      = (float*)d_ws;                      // [8192,2048]
    float* enc_hs  = zx     + (size_t)8192 * 2048;      // [8192,512]
    float* Wh_enc  = enc_hs + (size_t)8192 * 512;       // [8192,512]
    float* cstate  = Wh_enc + (size_t)8192 * 512;       // [32,512]
    float* hWs     = cstate + 32 * 512;                 // [32,512]
    float* energy  = hWs    + 32 * 512;                 // [32,256]
    float* zdec    = energy + 32 * 256;                 // [32,2048]
    float* logits  = zdec   + 32 * 2048;                // [32,32000]

    _Float16* src_emb16 = (_Float16*)(logits + (size_t)32 * 32000); // [8192,256]
    _Float16* enc_hs16  = src_emb16 + (size_t)8192 * 256;           // [8192,512]
    _Float16* enc_WxT   = enc_hs16  + (size_t)8192 * 512;           // [2048,256]
    _Float16* enc_WhT   = enc_WxT   + (size_t)2048 * 256;           // [2048,512]
    _Float16* attn_WhT  = enc_WhT   + (size_t)2048 * 512;           // [512,512]
    _Float16* attn_WsT  = attn_WhT  + (size_t)512 * 512;            // [512,512]
    _Float16* dec_WxT   = attn_WsT  + (size_t)512 * 512;            // [2048,768]
    _Float16* dec_WhT   = dec_WxT   + (size_t)2048 * 768;           // [2048,512]
    _Float16* out_WT    = dec_WhT   + (size_t)2048 * 512;           // [32000,1024]
    _Float16* hstate16  = out_WT    + (size_t)32000 * 1024;         // [32,512]
    _Float16* xbuf16    = hstate16  + 32 * 512;                     // [32,768]
    _Float16* hc16      = xbuf16    + 32 * 768;                     // [32,1024]

    dim3 tb(32, 8);
    // ---- one-time weight prep: f32 -> f16 transposed ----
    transpose_to_f16<<<dim3(  64,  8), tb, 0, stream>>>(enc_Wx,  enc_WxT,   256,  2048);
    transpose_to_f16<<<dim3(  64, 16), tb, 0, stream>>>(enc_Wh,  enc_WhT,   512,  2048);
    transpose_to_f16<<<dim3(  16, 16), tb, 0, stream>>>(attn_Wh, attn_WhT,  512,   512);
    transpose_to_f16<<<dim3(  16, 16), tb, 0, stream>>>(attn_Ws, attn_WsT,  512,   512);
    transpose_to_f16<<<dim3(  64, 24), tb, 0, stream>>>(dec_Wx,  dec_WxT,   768,  2048);
    transpose_to_f16<<<dim3(  64, 16), tb, 0, stream>>>(dec_Wh,  dec_WhT,   512,  2048);
    transpose_to_f16<<<dim3(1000, 32), tb, 0, stream>>>(out_W,   out_WT,   1024, 32000);
    gather_rows_f16<<<(8192 * 256) / 256, 256, 0, stream>>>(src, enc_embed,
                                                            src_emb16, 256, 8192 * 256);

    // 1) zx = src_emb @ enc_Wx + bx   (M=8192, K=256, N=2048)
    wmma_gemm16<<<(256 * 128) / 4, 128, 0, stream>>>(
        src_emb16, 256, enc_WxT, 256, enc_bx, zx, 2048, 256, 128, 256, 0);

    // 2) serial encoder scan (persistent single-WGP kernel, h/c in LDS)
    {
        size_t sm = (size_t)2 * 32 * HPAD * sizeof(_Float16)
                  + (size_t)32 * HPAD * sizeof(float);
        encoder_scan<<<1, 1024, sm, stream>>>(zx, enc_WhT, enc_bh, enc_hs,
                                              enc_hs16, hstate16, cstate, 256);
    }

    // 3) Wh_enc = enc_hs @ attn_Wh   (M=8192, K=512, N=512)
    wmma_gemm16<<<(256 * 32) / 4, 128, 0, stream>>>(
        enc_hs16, 512, attn_WhT, 512, nullptr, Wh_enc, 512, 256, 32, 512, 0);

    // 4) decoder: 64 steps
    for (int t = 0; t < 64; ++t) {
        // hWs = h @ attn_Ws   (M=32, K=512, N=512)
        wmma_gemm16<<<(1 * 32) / 4, 128, 0, stream>>>(
            hstate16, 512, attn_WsT, 512, nullptr, hWs, 512, 1, 32, 512, 0);
        attn_energy<<<1024, 256, 0, stream>>>(Wh_enc, hWs, attn_v, energy);
        softmax_ctx<<<32, 256, 0, stream>>>(energy, enc_hs, hc16, xbuf16);
        build_x<<<32, 256, 0, stream>>>(tgt, t, dec_embed, xbuf16);
        // z = x @ dec_Wx + bx   (M=32, K=768, N=2048)
        wmma_gemm16<<<(1 * 128) / 4, 128, 0, stream>>>(
            xbuf16, 768, dec_WxT, 768, dec_bx, zdec, 2048, 1, 128, 768, 0);
        // z += h @ dec_Wh + bh  (M=32, K=512, N=2048)
        wmma_gemm16<<<(1 * 128) / 4, 128, 0, stream>>>(
            hstate16, 512, dec_WhT, 512, dec_bh, zdec, 2048, 1, 128, 512, 1);
        dec_gates<<<(32 * 512) / 256, 256, 0, stream>>>(zdec, hstate16, cstate, hc16);
        // logits = [h, ctx] @ out_W + b   (M=32, K=1024, N=32000) — 134 GFLOP hot loop
        wmma_gemm16<<<(1 * 2000) / 4, 128, 0, stream>>>(
            hc16, 1024, out_WT, 1024, out_b, logits, 32000, 1, 2000, 1024, 0);
        log_softmax_out<<<32, 256, 0, stream>>>(logits, out, t);
    }
}
